// GemmaForCausalLM_27616639713598
// MI455X (gfx1250) — matrix operand order repaired and checked by reading, not compiled, not run on hardware
//
#include <hip/hip_runtime.h>
#include <hip/hip_bf16.h>
#include <math.h>

// ---------------- model constants ----------------
#define D_MODEL 2304
#define NHEADS  8
#define NKVH    4
#define HEADDIM 256
#define FFDIM   9216
#define QKVW    4096          // (H + 2*KVH) * HD
#define NLAYER  2
#define BATCH   2
#define SEQ     2048
#define MTOK    (BATCH*SEQ)   // 4096
#define WINSZ   1024
#define NEGINF  (-2.3819763e+38f)
#define RMSEPS  1e-6f

// ---------------- vector types (CDNA5 WMMA operands) ----------------
typedef __attribute__((ext_vector_type(16))) __bf16 v16bf;
typedef __attribute__((ext_vector_type(8)))  __bf16 v8bf;
typedef __attribute__((ext_vector_type(4)))  __bf16 v4bf;
typedef __attribute__((ext_vector_type(8)))  float  v8f;

__device__ __forceinline__ v16bf cat16(v8bf lo, v8bf hi) {
  return __builtin_shufflevector(lo, hi, 0,1,2,3,4,5,6,7,8,9,10,11,12,13,14,15);
}
__device__ __forceinline__ v4bf cvt4(float4 v) {
  v4bf r; r[0] = (__bf16)v.x; r[1] = (__bf16)v.y; r[2] = (__bf16)v.z; r[3] = (__bf16)v.w;
  return r;
}
__device__ __forceinline__ v16bf pack16(float4 a, float4 b, float4 c, float4 d) {
  v16bf r;
  r[0]=(__bf16)a.x; r[1]=(__bf16)a.y; r[2]=(__bf16)a.z;  r[3]=(__bf16)a.w;
  r[4]=(__bf16)b.x; r[5]=(__bf16)b.y; r[6]=(__bf16)b.z;  r[7]=(__bf16)b.w;
  r[8]=(__bf16)c.x; r[9]=(__bf16)c.y; r[10]=(__bf16)c.z; r[11]=(__bf16)c.w;
  r[12]=(__bf16)d.x;r[13]=(__bf16)d.y;r[14]=(__bf16)d.z; r[15]=(__bf16)d.w;
  return r;
}
__device__ __forceinline__ v8f wmma_bf16(v16bf a, v16bf b, v8f c) {
  return __builtin_amdgcn_wmma_f32_16x16x32_bf16(false, a, false, b, (short)0, c, false, false);
}
__device__ __forceinline__ float gelu_tanh(float x) {
  float x3 = x*x*x;
  return 0.5f*x*(1.0f + tanhf(0.7978845608028654f*(x + 0.044715f*x3)));
}
// gfx1250 async DMA global -> LDS (tracked by ASYNCcnt), 16B per active lane
__device__ __forceinline__ void async_copy_b128(unsigned lds_off, const float* gaddr) {
  asm volatile("global_load_async_to_lds_b128 %0, %1, off"
               :: "v"(lds_off), "v"(gaddr) : "memory");
}
__device__ __forceinline__ void wait_asynccnt0() {
  asm volatile("s_wait_asynccnt 0x0" ::: "memory");
}

// ---------------- embedding: h = embed[tok] * sqrt(D) ----------------
__global__ void embed_kernel(const int* __restrict__ tokens,
                             const float* __restrict__ emb,
                             float* __restrict__ h) {
  int t = blockIdx.x;
  int tok = tokens[t];
  const float* src = emb + (size_t)tok * D_MODEL;
  float* dst = h + (size_t)t * D_MODEL;
  for (int i = threadIdx.x; i < D_MODEL; i += blockDim.x)
    dst[i] = src[i] * 48.0f;                 // sqrt(2304) == 48 exactly
}

// ---------------- rmsnorm (optionally residual-add) ----------------
template<bool ADD>
__global__ void rmsnorm_kernel(const float* __restrict__ x,
                               const float* __restrict__ w,
                               float* __restrict__ y) {
  int row = blockIdx.x;
  const float* xr = x + (size_t)row * D_MODEL;
  int tid = threadIdx.x, lane = tid & 31, wv = tid >> 5;
  float ss = 0.0f;
  for (int i = tid; i < D_MODEL; i += 256) { float v = xr[i]; ss += v * v; }
  #pragma unroll
  for (int m = 16; m >= 1; m >>= 1) ss += __shfl_xor(ss, m, 32);
  __shared__ float red[8];
  __shared__ float stot;
  if (lane == 0) red[wv] = ss;
  __syncthreads();
  if (tid == 0) { float t = 0; for (int i = 0; i < 8; i++) t += red[i]; stot = t; }
  __syncthreads();
  float inv = rsqrtf(stot / (float)D_MODEL + RMSEPS);
  float* yr = y + (size_t)row * D_MODEL;
  for (int i = tid; i < D_MODEL; i += 256) {
    float v = xr[i] * inv * (1.0f + w[i]);
    if (ADD) yr[i] += v; else yr[i] = v;
  }
}

// ---------------- RoPE in-place on qkv buffer (+ q scaling) ----------------
__global__ void rope_kernel(float* __restrict__ qkv) {
  int t = blockIdx.x;                 // token
  int j = blockIdx.y;                 // 0..7 q heads, 8..11 k heads
  int i = threadIdx.x;                // pair index 0..127
  int pos = t & (SEQ - 1);
  float invf = __expf(-logf(10000.0f) * (float)(2 * i) / (float)HEADDIM);
  float ang = (float)pos * invf;
  float c = cosf(ang), s = sinf(ang);
  size_t base = (size_t)t * QKVW +
                (j < NHEADS ? (size_t)j * HEADDIM
                            : (size_t)NHEADS * HEADDIM + (size_t)(j - NHEADS) * HEADDIM);
  float scale = (j < NHEADS) ? 0.0625f : 1.0f;   // 256^-0.5
  float x1 = qkv[base + i], x2 = qkv[base + 128 + i];
  qkv[base + i]       = (x1 * c - x2 * s) * scale;
  qkv[base + 128 + i] = (x1 * s + x2 * c) * scale;
}

// ---------------- WMMA bf16 GEMM: C[M,N] = A[M,K] * B[K,N] ----------------
// Block tile 128x128, BK=32. 8 waves, each owns a 32x64 macro-tile (2x4 WMMA frags).
// A tile DMA'd fp32 global->LDS via GLOBAL_LOAD_ASYNC_TO_LDS_B128 (ASYNCcnt),
// converted to bf16 at fragment build; B staged via VGPRs (transpose+convert).
// mode 0: C = acc ; mode 1: C = gelu(acc) ; mode 2: C = C * acc
__global__ __launch_bounds__(256)
void gemm_kernel(const float* __restrict__ A, const float* __restrict__ Bw,
                 float* __restrict__ C, int M, int N, int K, int mode) {
  alignas(16) __shared__ float  Afp[2][128][36];   // fp32 A tile, pad 4 (36 KB)
  alignas(16) __shared__ __bf16 Bs [2][128][40];   // bf16 B^T tile [n][k] (20 KB)

  int tid = threadIdx.x, lane = tid & 31, wv = tid >> 5;
  int wm = wv >> 1, wn = wv & 1;
  int half = lane >> 4, nn = lane & 15;
  int m0 = blockIdx.y * 128, n0 = blockIdx.x * 128;
  int ar = tid >> 3,  ac = (tid & 7)  << 2;   // A: 4 rows, 32 apart
  int br = tid >> 5,  bc = (tid & 31) << 2;   // B: 4 rows, 8 apart

  v8f acc[2][4] = {};

  // ---- prologue: tile 0 ----
  #pragma unroll
  for (int i = 0; i < 4; i++) {
    unsigned lo = (unsigned)(uintptr_t)&Afp[0][ar + 32*i][ac];
    async_copy_b128(lo, A + (size_t)(m0 + ar + 32*i) * K + ac);
  }
  #pragma unroll
  for (int i = 0; i < 4; i++) {
    float4 v = *reinterpret_cast<const float4*>(Bw + (size_t)(br + 8*i) * N + n0 + bc);
    Bs[0][bc + 0][br + 8*i] = (__bf16)v.x;
    Bs[0][bc + 1][br + 8*i] = (__bf16)v.y;
    Bs[0][bc + 2][br + 8*i] = (__bf16)v.z;
    Bs[0][bc + 3][br + 8*i] = (__bf16)v.w;
  }
  wait_asynccnt0();
  __syncthreads();

  int ksteps = K >> 5;
  int cur = 0;
  for (int ks = 0; ks < ksteps; ks++) {
    int nxt = cur ^ 1;
    bool haveNext = (ks + 1 < ksteps);
    float4 bReg[4];
    if (haveNext) {
      int k0 = (ks + 1) << 5;
      // async DMA next A tile straight into LDS (overlaps with WMMA below)
      #pragma unroll
      for (int i = 0; i < 4; i++) {
        unsigned lo = (unsigned)(uintptr_t)&Afp[nxt][ar + 32*i][ac];
        async_copy_b128(lo, A + (size_t)(m0 + ar + 32*i) * K + k0 + ac);
      }
      #pragma unroll
      for (int i = 0; i < 4; i++)
        bReg[i] = *reinterpret_cast<const float4*>(Bw + (size_t)(k0 + br + 8*i) * N + n0 + bc);
      if (ks + 2 < ksteps)
        __builtin_prefetch(Bw + (size_t)((ks + 2) << 5) * N + n0 + bc, 0, 1);
    }

    // ---- fragments from LDS + WMMA ----
    v16bf af[2];
    #pragma unroll
    for (int a = 0; a < 2; a++) {
      const float* p = &Afp[cur][wm*32 + a*16 + nn][0];
      float4 q0 = *reinterpret_cast<const float4*>(p + half*8);
      float4 q1 = *reinterpret_cast<const float4*>(p + half*8 + 4);
      float4 q2 = *reinterpret_cast<const float4*>(p + 16 + half*8);
      float4 q3 = *reinterpret_cast<const float4*>(p + 16 + half*8 + 4);
      af[a] = pack16(q0, q1, q2, q3);
    }
    v16bf bfr[4];
    #pragma unroll
    for (int b2 = 0; b2 < 4; b2++) {
      const __bf16* p = &Bs[cur][wn*64 + b2*16 + nn][half * 8];
      bfr[b2] = cat16(*reinterpret_cast<const v8bf*>(p),
                      *reinterpret_cast<const v8bf*>(p + 16));
    }
    #pragma unroll
    for (int a = 0; a < 2; a++)
      #pragma unroll
      for (int b2 = 0; b2 < 4; b2++)
        acc[a][b2] = wmma_bf16(af[a], bfr[b2], acc[a][b2]);

    if (haveNext) {
      #pragma unroll
      for (int i = 0; i < 4; i++) {
        Bs[nxt][bc + 0][br + 8*i] = (__bf16)bReg[i].x;
        Bs[nxt][bc + 1][br + 8*i] = (__bf16)bReg[i].y;
        Bs[nxt][bc + 2][br + 8*i] = (__bf16)bReg[i].z;
        Bs[nxt][bc + 3][br + 8*i] = (__bf16)bReg[i].w;
      }
    }
    wait_asynccnt0();     // async writes to LDS[nxt] must land before barrier
    __syncthreads();
    cur = nxt;
  }

  // ---- epilogue ----
  #pragma unroll
  for (int a = 0; a < 2; a++)
    #pragma unroll
    for (int b2 = 0; b2 < 4; b2++) {
      int n = n0 + wn*64 + b2*16 + nn;
      #pragma unroll
      for (int r = 0; r < 8; r++) {
        int m = m0 + wm*32 + a*16 + half*8 + r;
        size_t idx = (size_t)m * N + n;
        float v = acc[a][b2][r];
        if (mode == 1)      v = gelu_tanh(v);
        else if (mode == 2) v = C[idx] * v;
        C[idx] = v;
      }
    }
}

// ---------------- WMMA flash attention ----------------
// grid (S/64, H, B), 128 threads (4 waves, 16 q-rows/wave), 32-key blocks
__global__ __launch_bounds__(128)
void attn_kernel(const float* __restrict__ qkv, float* __restrict__ out, int win) {
  alignas(16) __shared__ __bf16 KS[32][HEADDIM + 8];   // K tile [key][hd]
  alignas(16) __shared__ __bf16 VS[HEADDIM][32 + 8];   // V tile transposed [hd][key]
  alignas(16) __shared__ __bf16 PS[4][16][32 + 8];     // per-wave P tile [row][key]

  int q0 = blockIdx.x * 64;
  int h  = blockIdx.y;
  int b  = blockIdx.z;
  int kvh = h >> 1;                                    // H/KVH == 2
  int tid = threadIdx.x, lane = tid & 31, wv = tid >> 5;
  int half = lane >> 4, nn = lane & 15;

  // Q fragments straight from global (A-matrix layout), rope+scale already applied
  v16bf qf[8];
  {
    int qrow = b * SEQ + q0 + wv * 16 + nn;
    const float* qb = qkv + (size_t)qrow * QKVW + (size_t)h * HEADDIM;
    #pragma unroll
    for (int kk = 0; kk < 8; kk++) {
      int c0 = kk * 32 + half * 8;
      int c1 = kk * 32 + 16 + half * 8;
      float4 f0 = *reinterpret_cast<const float4*>(qb + c0);
      float4 f1 = *reinterpret_cast<const float4*>(qb + c0 + 4);
      float4 f2 = *reinterpret_cast<const float4*>(qb + c1);
      float4 f3 = *reinterpret_cast<const float4*>(qb + c1 + 4);
      qf[kk] = pack16(f0, f1, f2, f3);
    }
  }

  float mrow[8], lrow[8];
  v8f accO[16] = {};
  #pragma unroll
  for (int r = 0; r < 8; r++) { mrow[r] = NEGINF; lrow[r] = 0.0f; }

  int kb_hi = (q0 + 63) >> 5;
  int kb_lo = 0;
  if (win < SEQ) { int lowk = q0 - (win - 1); if (lowk > 0) kb_lo = lowk >> 5; }

  for (int kb = kb_lo; kb <= kb_hi; kb++) {
    __syncthreads();
    // stage K/V tiles: 32 keys x 256 hd, fp32 -> bf16
    #pragma unroll 4
    for (int it = 0; it < 16; it++) {
      int slot = it * 128 + tid;
      int r = slot >> 6;
      int c4 = (slot & 63) << 2;
      size_t trow = (size_t)(b * SEQ + kb * 32 + r) * QKVW;
      float4 kv = *reinterpret_cast<const float4*>(qkv + trow + 2048 + kvh * HEADDIM + c4);
      *reinterpret_cast<v4bf*>(&KS[r][c4]) = cvt4(kv);
      float4 vvv = *reinterpret_cast<const float4*>(qkv + trow + 3072 + kvh * HEADDIM + c4);
      VS[c4 + 0][r] = (__bf16)vvv.x;
      VS[c4 + 1][r] = (__bf16)vvv.y;
      VS[c4 + 2][r] = (__bf16)vvv.z;
      VS[c4 + 3][r] = (__bf16)vvv.w;
    }
    __syncthreads();

    // S = Q K^T for 2 n-tiles of 16 keys each
    float p[2][8];
    float rowmax[8];
    #pragma unroll
    for (int r = 0; r < 8; r++) rowmax[r] = NEGINF;

    #pragma unroll
    for (int nt = 0; nt < 2; nt++) {
      v8f s = {};
      #pragma unroll
      for (int kk = 0; kk < 8; kk++) {
        const __bf16* pb = &KS[nt * 16 + nn][kk * 32 + half * 8];
        v16bf bfrag = cat16(*reinterpret_cast<const v8bf*>(pb),
                            *reinterpret_cast<const v8bf*>(pb + 16));
        s = wmma_bf16(qf[kk], bfrag, s);
      }
      int kj = kb * 32 + nt * 16 + nn;
      #pragma unroll
      for (int r = 0; r < 8; r++) {
        int qi = q0 + wv * 16 + half * 8 + r;
        float sv = s[r];
        bool valid = (kj <= qi) && ((qi - kj) <= (win - 1));
        sv = valid ? tanhf(sv * 0.02f) * 50.0f : NEGINF;
        p[nt][r] = sv;
        rowmax[r] = fmaxf(rowmax[r], sv);
      }
    }
    // reduce row max across the 16 lanes of each half-wave
    #pragma unroll
    for (int m = 8; m >= 1; m >>= 1)
      #pragma unroll
      for (int r = 0; r < 8; r++)
        rowmax[r] = fmaxf(rowmax[r], __shfl_xor(rowmax[r], m, 32));

    float alpha[8], rs[8];
    #pragma unroll
    for (int r = 0; r < 8; r++) {
      float mnew = fmaxf(mrow[r], rowmax[r]);
      alpha[r] = __expf(mrow[r] - mnew);
      mrow[r] = mnew;
      rs[r] = 0.0f;
    }
    #pragma unroll
    for (int nt = 0; nt < 2; nt++)
      #pragma unroll
      for (int r = 0; r < 8; r++) {
        float sv = p[nt][r];
        float pv = (sv > -1e37f) ? __expf(sv - mrow[r]) : 0.0f;
        p[nt][r] = pv;
        rs[r] += pv;
      }
    #pragma unroll
    for (int m = 8; m >= 1; m >>= 1)
      #pragma unroll
      for (int r = 0; r < 8; r++)
        rs[r] += __shfl_xor(rs[r], m, 32);
    #pragma unroll
    for (int r = 0; r < 8; r++) lrow[r] = lrow[r] * alpha[r] + rs[r];
    #pragma unroll
    for (int t = 0; t < 16; t++)
      #pragma unroll
      for (int r = 0; r < 8; r++) accO[t][r] *= alpha[r];

    // C-layout -> A-layout via per-wave LDS round trip
    #pragma unroll
    for (int nt = 0; nt < 2; nt++)
      #pragma unroll
      for (int r = 0; r < 8; r++)
        PS[wv][half * 8 + r][nt * 16 + nn] = (__bf16)p[nt][r];

    const __bf16* pp = &PS[wv][nn][half * 8];
    v16bf pfrag = cat16(*reinterpret_cast<const v8bf*>(pp),
                        *reinterpret_cast<const v8bf*>(pp + 16));
    #pragma unroll
    for (int nt2 = 0; nt2 < 16; nt2++) {
      const __bf16* vb = &VS[nt2 * 16 + nn][half * 8];
      v16bf vfrag = cat16(*reinterpret_cast<const v8bf*>(vb),
                          *reinterpret_cast<const v8bf*>(vb + 16));
      accO[nt2] = wmma_bf16(pfrag, vfrag, accO[nt2]);
    }
  }

  // epilogue: out[b,s, h*HD + hd] = O / l
  #pragma unroll
  for (int r = 0; r < 8; r++) {
    float inv = 1.0f / lrow[r];
    size_t row = (size_t)(b * SEQ + q0 + wv * 16 + half * 8 + r) * (NHEADS * HEADDIM)
               + (size_t)h * HEADDIM;
    #pragma unroll
    for (int t = 0; t < 16; t++)
      out[row + t * 16 + nn] = accO[t][r] * inv;
  }
}

// ---------------- host orchestration ----------------
extern "C" void kernel_launch(void* const* d_in, const int* in_sizes, int n_in,
                              void* d_out, int out_size, void* d_ws, size_t ws_size,
                              hipStream_t stream) {
  (void)in_sizes; (void)n_in; (void)out_size; (void)ws_size;
  const int*   tokens   = (const int*)  d_in[0];
  const float* emb      = (const float*)d_in[1];
  const float* qkv_w    = (const float*)d_in[2];
  const float* o_w      = (const float*)d_in[3];
  const float* gate_w   = (const float*)d_in[4];
  const float* up_w     = (const float*)d_in[5];
  const float* down_w   = (const float*)d_in[6];
  const float* ln_in    = (const float*)d_in[7];
  const float* ln_pa    = (const float*)d_in[8];
  const float* ln_pf    = (const float*)d_in[9];
  const float* ln_pff   = (const float*)d_in[10];
  const float* ln_final = (const float*)d_in[11];
  float* out = (float*)d_out;

  float* hbuf  = (float*)d_ws;
  float* xb    = hbuf  + (size_t)MTOK * D_MODEL;
  float* qkvb  = xb    + (size_t)MTOK * D_MODEL;
  float* attnb = qkvb  + (size_t)MTOK * QKVW;
  float* bigb  = attnb + (size_t)MTOK * (NHEADS * HEADDIM);

  embed_kernel<<<MTOK, 256, 0, stream>>>(tokens, emb, hbuf);

  for (int l = 0; l < NLAYER; l++) {
    const float* qw = qkv_w  + (size_t)l * D_MODEL * QKVW;
    const float* ow = o_w    + (size_t)l * (NHEADS * HEADDIM) * D_MODEL;
    const float* gw = gate_w + (size_t)l * D_MODEL * FFDIM;
    const float* uw = up_w   + (size_t)l * D_MODEL * FFDIM;
    const float* dw = down_w + (size_t)l * FFDIM * D_MODEL;

    // attention block
    rmsnorm_kernel<false><<<MTOK, 256, 0, stream>>>(hbuf, ln_in + l * D_MODEL, xb);
    gemm_kernel<<<dim3(QKVW/128, MTOK/128), 256, 0, stream>>>(xb, qw, qkvb,
                                                              MTOK, QKVW, D_MODEL, 0);
    rope_kernel<<<dim3(MTOK, NHEADS + NKVH), 128, 0, stream>>>(qkvb);
    int win = (l % 2 == 0) ? WINSZ : SEQ;
    attn_kernel<<<dim3(SEQ/64, NHEADS, BATCH), 128, 0, stream>>>(qkvb, attnb, win);
    gemm_kernel<<<dim3(D_MODEL/128, MTOK/128), 256, 0, stream>>>(attnb, ow, xb,
                                                                 MTOK, D_MODEL,
                                                                 NHEADS * HEADDIM, 0);
    rmsnorm_kernel<true><<<MTOK, 256, 0, stream>>>(xb, ln_pa + l * D_MODEL, hbuf);

    // MLP block
    rmsnorm_kernel<false><<<MTOK, 256, 0, stream>>>(hbuf, ln_pf + l * D_MODEL, xb);
    gemm_kernel<<<dim3(FFDIM/128, MTOK/128), 256, 0, stream>>>(xb, gw, bigb,
                                                               MTOK, FFDIM, D_MODEL, 1);
    gemm_kernel<<<dim3(FFDIM/128, MTOK/128), 256, 0, stream>>>(xb, uw, bigb,
                                                               MTOK, FFDIM, D_MODEL, 2);
    gemm_kernel<<<dim3(D_MODEL/128, MTOK/128), 256, 0, stream>>>(bigb, dw, xb,
                                                                 MTOK, D_MODEL, FFDIM, 0);
    rmsnorm_kernel<true><<<MTOK, 256, 0, stream>>>(xb, ln_pff + l * D_MODEL, hbuf);
  }

  rmsnorm_kernel<false><<<MTOK, 256, 0, stream>>>(hbuf, ln_final, out);
}
